// MGVDModel_52776558133561
// MI455X (gfx1250) — compile-verified, matrix operand order checked by reading, not run on hardware
//
#include <hip/hip_runtime.h>
#include <hip/hip_bf16.h>
#include <math.h>

// ---------------------------------------------------------------------------
// Multi-graph GNN (2x GIN + 2x GAT) -> pool -> SE gate -> MLP classifier.
// Dense GEMMs: v_wmma_f32_16x16x32_f16, A loaded as contiguous v8f runs per
// the ISA 7.12.2 fragment layout, weights pre-transposed to f16 col-major.
// All fragment loads for a K-step are issued before the WMMA chain so the
// scheduler can clause them and stagger s_wait_loadcnt across the 4 WMMAs.
// Irregular edge aggregation: native f32 global atomics (fits 192MB L2).
// ---------------------------------------------------------------------------

#define NNODE 50000
#define NEDGE 200000
#define NGRAPH 256
#define DIN 64
#define HID 128
#define NHEADS 4
#define NCLS 7
#define GATW (NHEADS * HID) // 512
#define TN 4                // N-tiles per wave in the WMMA GEMM

typedef __attribute__((ext_vector_type(16))) _Float16 v16h;
typedef __attribute__((ext_vector_type(8)))  _Float16 v8h;
typedef __attribute__((ext_vector_type(8)))  float    v8f;

// ---------------------------------------------------------------------------
// Weight prep: Wt[n*K + k] = (f16) W[k*Nc + n]   (f32 row-major -> f16 col-major)
// ---------------------------------------------------------------------------
__global__ void weight_to_f16t_kernel(const float* __restrict__ W,
                                      _Float16* __restrict__ Wt, int K, int Nc) {
    int i = blockIdx.x * blockDim.x + threadIdx.x;
    if (i >= K * Nc) return;
    int k = i / Nc, n = i % Nc;
    Wt[(size_t)n * K + k] = (_Float16)W[i];
}

// ---------------------------------------------------------------------------
// WMMA GEMM: C[M,Nc] = act(A[M,K] @ B[K,Nc] + bias[Nc]), B given as f16 Bt[Nc,K].
// One wave computes a 16 x (TN*16) strip. K stepped by 32 (K % 32 == 0 at all
// call sites: 64/128/256/512). Fragment K-mapping per CDNA5 ISA 7.12.2:
// elem e<8 -> k=half*8+e, e>=8 -> k=16+half*8+(e-8); both runs contiguous.
// act: 0=none 1=relu 2=sigmoid
// ---------------------------------------------------------------------------
__global__ void gemm_wmma_f16_kernel(const float* __restrict__ A,
                                     const _Float16* __restrict__ Bt,
                                     const float* __restrict__ bias,
                                     float* __restrict__ C,
                                     int M, int Nc, int K, int act) {
    int wave = (int)((blockIdx.x * (long long)blockDim.x + threadIdx.x) >> 5);
    int lane = threadIdx.x & 31;
    int mt  = (M + 15) >> 4;
    int ntg = (Nc + TN * 16 - 1) / (TN * 16);
    if (wave >= mt * ntg) return;
    int mtile = wave % mt;
    int ng    = wave / mt;
    int half  = lane >> 4;
    int r     = lane & 15;

    int arow = mtile * 16 + r;
    const float* Ap = A + (size_t)((arow < M) ? arow : (M - 1)) * K;

    const _Float16* Bp[TN];
#pragma unroll
    for (int t = 0; t < TN; ++t) {
        int bc = ng * (TN * 16) + t * 16 + r;
        // Clamp OOB columns: their products land only in OOB output columns,
        // which the guarded epilogue never stores.
        Bp[t] = Bt + (size_t)((bc < Nc) ? bc : (Nc - 1)) * K;
    }

    v8f zero = {};
    v8f acc[TN];
#pragma unroll
    for (int t = 0; t < TN; ++t) acc[t] = zero;

    for (int k0 = 0; k0 < K; k0 += 32) {
        int kb = k0 + half * 8;
        // Issue ALL loads for this K-step first (one clause, staggered waits).
        v8f alo = *(const v8f*)(Ap + kb);        // k = kb .. kb+7
        v8f ahi = *(const v8f*)(Ap + kb + 16);   // k = kb+16 .. kb+23
        v8h bl[TN], bh[TN];
#pragma unroll
        for (int t = 0; t < TN; ++t) {
            bl[t] = *(const v8h*)(Bp[t] + kb);
            bh[t] = *(const v8h*)(Bp[t] + kb + 16);
        }
        v8h al = __builtin_convertvector(alo, v8h);
        v8h ah = __builtin_convertvector(ahi, v8h);
        v16h af = __builtin_shufflevector(al, ah, 0, 1, 2, 3, 4, 5, 6, 7,
                                          8, 9, 10, 11, 12, 13, 14, 15);
#pragma unroll
        for (int t = 0; t < TN; ++t) {
            v16h bf = __builtin_shufflevector(bl[t], bh[t], 0, 1, 2, 3, 4, 5, 6, 7,
                                              8, 9, 10, 11, 12, 13, 14, 15);
            acc[t] = __builtin_amdgcn_wmma_f32_16x16x32_f16(
                false, af, false, bf, (short)0, acc[t], false, false);
        }
    }

#pragma unroll
    for (int t = 0; t < TN; ++t) {
        int no = ng * (TN * 16) + t * 16 + r;
#pragma unroll
        for (int rr = 0; rr < 8; ++rr) {
            int mo = mtile * 16 + rr + half * 8;
            if (mo < M && no < Nc) {
                float v = acc[t][rr] + (bias ? bias[no] : 0.0f);
                if (act == 1)      v = fmaxf(v, 0.0f);
                else if (act == 2) v = 1.0f / (1.0f + __expf(-v));
                C[(size_t)mo * Nc + no] = v;
            }
        }
    }
}

// --------------------------- elementwise / scatter -------------------------
__global__ void fill_f32_kernel(float* __restrict__ p, float v, long long n) {
    long long i = blockIdx.x * (long long)blockDim.x + threadIdx.x;
    if (i < n) p[i] = v;
}

// GIN aggregation: out[dst] += x[src] (out pre-initialized with x => x + sum)
__global__ void scatter_add_kernel(const float* __restrict__ x,
                                   const int* __restrict__ src,
                                   const int* __restrict__ dst,
                                   float* __restrict__ out, int E, int dim) {
    long long i = blockIdx.x * (long long)blockDim.x + threadIdx.x;
    if (i >= (long long)E * dim) return;
    int e = (int)(i / dim);
    int f = (int)(i % dim);
    atomicAdd(&out[(size_t)dst[e] * dim + f], x[(size_t)src[e] * dim + f]);
}

// per-(node,head) attention scores: s = sum_k H[n,h,k]*att[h,k]
__global__ void gat_scores_kernel(const float* __restrict__ H,
                                  const float* __restrict__ att_src,
                                  const float* __restrict__ att_dst,
                                  float* __restrict__ ssrc,
                                  float* __restrict__ sdst, int Nn) {
    int i = blockIdx.x * blockDim.x + threadIdx.x;
    if (i >= Nn * NHEADS) return;
    int node = i >> 2, hd = i & 3;
    const float* hp = H + (size_t)node * GATW + hd * HID;
    const float* as = att_src + hd * HID;
    const float* ad = att_dst + hd * HID;
    float s1 = 0.f, s2 = 0.f;
    for (int k = 0; k < HID; ++k) { float hv = hp[k]; s1 += hv * as[k]; s2 += hv * ad[k]; }
    ssrc[i] = s1;
    sdst[i] = s2;
}

__device__ inline void atomicMaxFloat(float* addr, float val) {
    unsigned int* u = (unsigned int*)addr;
    unsigned int old = *u;
    while (__uint_as_float(old) < val) {
        unsigned int assumed = old;
        old = atomicCAS(u, assumed, __float_as_uint(val));
        if (old == assumed) break;
    }
}

// leaky-relu edge score (incl. self loops at e>=E) + segment max over dst
__global__ void gat_edge_score_kernel(const int* __restrict__ src,
                                      const int* __restrict__ dst,
                                      const float* __restrict__ ssrc,
                                      const float* __restrict__ sdst,
                                      float* __restrict__ a,
                                      float* __restrict__ mbuf, int E, int Nn) {
    int i = blockIdx.x * blockDim.x + threadIdx.x;
    if (i >= (E + Nn) * NHEADS) return;
    int e = i >> 2, hd = i & 3;
    int s = (e < E) ? src[e] : (e - E);
    int d = (e < E) ? dst[e] : (e - E);
    float v = ssrc[s * NHEADS + hd] + sdst[d * NHEADS + hd];
    v = (v > 0.f) ? v : 0.2f * v;
    a[i] = v;
    atomicMaxFloat(&mbuf[d * NHEADS + hd], v);
}

// e = exp(a - m[dst]) (stable softmax) + segment-sum denominator
__global__ void gat_exp_kernel(const int* __restrict__ dst,
                               const float* __restrict__ mbuf,
                               float* __restrict__ a,
                               float* __restrict__ denom, int E, int Nn) {
    int i = blockIdx.x * blockDim.x + threadIdx.x;
    if (i >= (E + Nn) * NHEADS) return;
    int e = i >> 2, hd = i & 3;
    int d = (e < E) ? dst[e] : (e - E);
    float ev = __expf(a[i] - mbuf[d * NHEADS + hd]);
    a[i] = ev;
    atomicAdd(&denom[d * NHEADS + hd], ev);
}

// out[dst, h, :] += alpha * H[src, h, :]
__global__ void gat_aggregate_kernel(const int* __restrict__ src,
                                     const int* __restrict__ dst,
                                     const float* __restrict__ a,
                                     const float* __restrict__ denom,
                                     const float* __restrict__ H,
                                     float* __restrict__ out, int E, int Nn) {
    long long i = blockIdx.x * (long long)blockDim.x + threadIdx.x;
    if (i >= (long long)(E + Nn) * GATW) return;
    int e = (int)(i >> 9);      // GATW == 512
    int f = (int)(i & 511);
    int hd = f >> 7;
    int s = (e < E) ? src[e] : (e - E);
    int d = (e < E) ? dst[e] : (e - E);
    float alpha = a[(size_t)e * NHEADS + hd] / (denom[d * NHEADS + hd] + 1e-16f);
    atomicAdd(&out[(size_t)d * GATW + f], alpha * H[(size_t)s * GATW + f]);
}

// head mean + bias + relu -> [N, HID]
__global__ void gat_mean_kernel(const float* __restrict__ out,
                                const float* __restrict__ b,
                                float* __restrict__ x2, int Nn) {
    int i = blockIdx.x * blockDim.x + threadIdx.x;
    if (i >= Nn * HID) return;
    int node = i >> 7, k = i & 127;
    const float* op = out + (size_t)node * GATW;
    float v = 0.25f * (op[k] + op[HID + k] + op[2 * HID + k] + op[3 * HID + k]) + b[k];
    x2[i] = fmaxf(v, 0.0f);
}

// ------------------------------- pooling -----------------------------------
__global__ void pool_count_kernel(const int* __restrict__ batch,
                                  float* __restrict__ cnt, int Nn) {
    int i = blockIdx.x * blockDim.x + threadIdx.x;
    if (i < Nn) atomicAdd(&cnt[batch[i]], 1.0f);
}

__global__ void pool_sum_kernel(const float* __restrict__ x,
                                const int* __restrict__ batch,
                                float* __restrict__ c, int col_off, int Nn) {
    int i = blockIdx.x * blockDim.x + threadIdx.x;
    if (i >= Nn * HID) return;
    int node = i >> 7, k = i & 127;
    atomicAdd(&c[(size_t)batch[node] * (4 * HID) + col_off + k], x[i]);
}

__global__ void pool_div_kernel(float* __restrict__ c,
                                const float* __restrict__ cnt, int col_off) {
    int i = blockIdx.x * blockDim.x + threadIdx.x;
    if (i >= NGRAPH * HID) return;
    int g = i >> 7, k = i & 127;
    c[(size_t)g * (4 * HID) + col_off + k] /= fmaxf(cnt[g], 1.0f);
}

// SE gate: sig = c * sig (in place)
__global__ void se_mul_kernel(const float* __restrict__ c,
                              float* __restrict__ sig, long long n) {
    long long i = blockIdx.x * (long long)blockDim.x + threadIdx.x;
    if (i < n) sig[i] = c[i] * sig[i];
}

// ---------------------------------------------------------------------------
extern "C" void kernel_launch(void* const* d_in, const int* in_sizes, int n_in,
                              void* d_out, int out_size, void* d_ws, size_t ws_size,
                              hipStream_t stream) {
    (void)in_sizes; (void)n_in; (void)out_size; (void)ws_size;

    // ---- input layout: setup_inputs() insertion order; params flattened in
    // insertion order (ast, dfg, cfg, pdg, se1, se2, cl1, cl2, cl3) ----
    const float* gx[4];  const int* gei[4];  const int* gbatch[4];
    for (int g = 0; g < 4; ++g) {  // dict graph order: ast, cfg, dfg, pdg
        gx[g]     = (const float*)d_in[3 * g + 0];
        gei[g]    = (const int*)  d_in[3 * g + 1];
        gbatch[g] = (const int*)  d_in[3 * g + 2];
    }
    int p = 12;
    const float* gin_p[2][8];   // [w1,b1,w2,b2] x {c1,c2} for ast, dfg
    for (int b = 0; b < 2; ++b)
        for (int j = 0; j < 8; ++j) gin_p[b][j] = (const float*)d_in[p++];
    const float* gat_p[2][8];   // [W,att_src,att_dst,b] x {c1,c2} for cfg, pdg
    for (int b = 0; b < 2; ++b)
        for (int j = 0; j < 8; ++j) gat_p[b][j] = (const float*)d_in[p++];
    const float* se1w = (const float*)d_in[p++]; const float* se1b = (const float*)d_in[p++];
    const float* se2w = (const float*)d_in[p++]; const float* se2b = (const float*)d_in[p++];
    const float* cl1w = (const float*)d_in[p++]; const float* cl1b = (const float*)d_in[p++];
    const float* cl2w = (const float*)d_in[p++]; const float* cl2b = (const float*)d_in[p++];
    const float* cl3w = (const float*)d_in[p++]; const float* cl3b = (const float*)d_in[p++];

    // ---- workspace arena (floats) ----
    float* ws = (float*)d_ws;
    size_t o = 0;
    float* BIG0 = ws + o; o += (size_t)NNODE * GATW;            // GAT H / GIN hidden
    float* BIG1 = ws + o; o += (size_t)NNODE * GATW;            // GAT accum / GIN out
    float* X2   = ws + o; o += (size_t)NNODE * HID;             // layer-1 output
    float* TMP  = ws + o; o += (size_t)NNODE * HID;             // GIN (x+agg) / GAT out2
    float* EDG  = ws + o; o += (size_t)(NEDGE + NNODE) * NHEADS;
    float* SSRC = ws + o; o += (size_t)NNODE * NHEADS;
    float* SDST = ws + o; o += (size_t)NNODE * NHEADS;
    float* MBUF = ws + o; o += (size_t)NNODE * NHEADS;
    float* DEN  = ws + o; o += (size_t)NNODE * NHEADS;
    float* CNT  = ws + o; o += NGRAPH;
    float* CCAT = ws + o; o += (size_t)NGRAPH * 4 * HID;
    float* SEH  = ws + o; o += (size_t)NGRAPH * HID;
    float* SEO  = ws + o; o += (size_t)NGRAPH * 4 * HID;
    float* H1C  = ws + o; o += (size_t)NGRAPH * 2 * HID;
    float* H2C  = ws + o; o += (size_t)NGRAPH * HID;
    _Float16* WT = (_Float16*)(ws + o); o += 512 * 256 / 2 + 64; // f16 weight scratch (max 512x256)

    const int TB = 256;
    auto blk = [](long long n, int b) { return (unsigned)((n + b - 1) / b); };

    auto gemm = [&](const float* A, const float* W, const float* bias, float* C,
                    int M, int Nc, int K, int act) {
        weight_to_f16t_kernel<<<blk((long long)K * Nc, TB), TB, 0, stream>>>(W, WT, K, Nc);
        long long waves = (long long)((M + 15) / 16) * ((Nc + TN * 16 - 1) / (TN * 16));
        gemm_wmma_f16_kernel<<<blk(waves * 32, TB), TB, 0, stream>>>(A, WT, bias, C, M, Nc, K, act);
    };

    // GIN layer: out = relu( relu((x + sum_nbr x) @ w1 + b1) @ w2 + b2 )
    auto gin_layer = [&](const float* x, int din, const float* const* pp, int pbase,
                         const int* src, const int* dst, float* out) {
        hipMemcpyAsync(TMP, x, (size_t)NNODE * din * sizeof(float),
                       hipMemcpyDeviceToDevice, stream);
        scatter_add_kernel<<<blk((long long)NEDGE * din, TB), TB, 0, stream>>>(
            x, src, dst, TMP, NEDGE, din);
        gemm(TMP, pp[pbase + 0], pp[pbase + 1], BIG0, NNODE, HID, din, 1);
        gemm(BIG0, pp[pbase + 2], pp[pbase + 3], out, NNODE, HID, HID, 1);
    };

    // GAT layer (concat=False head mean, self loops, stable segment softmax)
    auto gat_layer = [&](const float* x, int din, const float* W, const float* asc,
                         const float* adc, const float* bb,
                         const int* src, const int* dst, float* out) {
        gemm(x, W, nullptr, BIG0, NNODE, GATW, din, 0);
        gat_scores_kernel<<<blk((long long)NNODE * NHEADS, TB), TB, 0, stream>>>(
            BIG0, asc, adc, SSRC, SDST, NNODE);
        fill_f32_kernel<<<blk((long long)NNODE * NHEADS, TB), TB, 0, stream>>>(
            MBUF, -1e30f, (long long)NNODE * NHEADS);
        gat_edge_score_kernel<<<blk((long long)(NEDGE + NNODE) * NHEADS, TB), TB, 0, stream>>>(
            src, dst, SSRC, SDST, EDG, MBUF, NEDGE, NNODE);
        hipMemsetAsync(DEN, 0, (size_t)NNODE * NHEADS * sizeof(float), stream);
        gat_exp_kernel<<<blk((long long)(NEDGE + NNODE) * NHEADS, TB), TB, 0, stream>>>(
            dst, MBUF, EDG, DEN, NEDGE, NNODE);
        hipMemsetAsync(BIG1, 0, (size_t)NNODE * GATW * sizeof(float), stream);
        gat_aggregate_kernel<<<blk((long long)(NEDGE + NNODE) * GATW, TB), TB, 0, stream>>>(
            src, dst, EDG, DEN, BIG0, BIG1, NEDGE, NNODE);
        gat_mean_kernel<<<blk((long long)NNODE * HID, TB), TB, 0, stream>>>(
            BIG1, bb, out, NNODE);
    };

    auto pool = [&](const float* h, const int* batch, int col_off) {
        hipMemsetAsync(CNT, 0, NGRAPH * sizeof(float), stream);
        pool_count_kernel<<<blk(NNODE, TB), TB, 0, stream>>>(batch, CNT, NNODE);
        pool_sum_kernel<<<blk((long long)NNODE * HID, TB), TB, 0, stream>>>(
            h, batch, CCAT, col_off, NNODE);
        pool_div_kernel<<<blk((long long)NGRAPH * HID, TB), TB, 0, stream>>>(
            CCAT, CNT, col_off);
    };

    hipMemsetAsync(CCAT, 0, (size_t)NGRAPH * 4 * HID * sizeof(float), stream);

    // ---- branches; concat column order: [ast, cfg, dfg, pdg] ----
    { // ast (GIN), cols 0
        const int* src = gei[0]; const int* dst = gei[0] + NEDGE;
        gin_layer(gx[0], DIN, gin_p[0], 0, src, dst, X2);
        gin_layer(X2, HID, gin_p[0], 4, src, dst, BIG1);
        pool(BIG1, gbatch[0], 0 * HID);
    }
    { // cfg (GAT), cols 128
        const int* src = gei[1]; const int* dst = gei[1] + NEDGE;
        gat_layer(gx[1], DIN, gat_p[0][0], gat_p[0][1], gat_p[0][2], gat_p[0][3], src, dst, X2);
        gat_layer(X2, HID, gat_p[0][4], gat_p[0][5], gat_p[0][6], gat_p[0][7], src, dst, TMP);
        pool(TMP, gbatch[1], 1 * HID);
    }
    { // dfg (GIN), cols 256
        const int* src = gei[2]; const int* dst = gei[2] + NEDGE;
        gin_layer(gx[2], DIN, gin_p[1], 0, src, dst, X2);
        gin_layer(X2, HID, gin_p[1], 4, src, dst, BIG1);
        pool(BIG1, gbatch[2], 2 * HID);
    }
    { // pdg (GAT), cols 384
        const int* src = gei[3]; const int* dst = gei[3] + NEDGE;
        gat_layer(gx[3], DIN, gat_p[1][0], gat_p[1][1], gat_p[1][2], gat_p[1][3], src, dst, X2);
        gat_layer(X2, HID, gat_p[1][4], gat_p[1][5], gat_p[1][6], gat_p[1][7], src, dst, TMP);
        pool(TMP, gbatch[3], 3 * HID);
    }

    // ---- SE gate + classifier ----
    gemm(CCAT, se1w, se1b, SEH, NGRAPH, HID, 4 * HID, 1);       // relu
    gemm(SEH, se2w, se2b, SEO, NGRAPH, 4 * HID, HID, 2);        // sigmoid
    se_mul_kernel<<<blk((long long)NGRAPH * 4 * HID, TB), TB, 0, stream>>>(
        CCAT, SEO, (long long)NGRAPH * 4 * HID);
    gemm(SEO, cl1w, cl1b, H1C, NGRAPH, 2 * HID, 4 * HID, 1);    // relu
    gemm(H1C, cl2w, cl2b, H2C, NGRAPH, HID, 2 * HID, 1);        // relu
    gemm(H2C, cl3w, cl3b, (float*)d_out, NGRAPH, NCLS, HID, 0); // logits
}
